// LSTMFeatureExtractor_39067022524824
// MI455X (gfx1250) — matrix operand order repaired and checked by reading, not compiled, not run on hardware
//
#include <hip/hip_runtime.h>
#include <cmath>

typedef __bf16 bf16;
typedef __attribute__((ext_vector_type(16))) __bf16 v16bf;
typedef __attribute__((ext_vector_type(8)))  float  v8f;

#define B_   256
#define T_   128
#define D_   64
#define H_   512
#define F_   256
#define G4_  2048          // 4*H
#define KT_  18            // (D+H)/32 k-steps
#define FRAG 512           // elems per fragment: 32 lanes * 16 bf16 (1024 bytes)

// ---- fragment load: 32 contiguous bytes per lane -> v16bf (two b128 loads) ----
static __device__ __forceinline__ v16bf load_frag(const bf16* p) {
    union { uint4 u[2]; v16bf v; } t;
    const uint4* q = (const uint4*)p;
    t.u[0] = q[0];
    t.u[1] = q[1];
    return t.v;
}

static __device__ __forceinline__ float sigmoidf_(float x) {
    return 1.0f / (1.0f + __expf(-x));
}

// =====================  prep: pack [Wx;Wh] into B-fragments  =====================
// B layout (16-bit, 32x16): elem e => K = 8*(lane>>4) + e + (e>=8 ? 8 : 0) (lanes 0-15: K 0..7,16..23; 16-31: K 8..15,24..31); N = lane&15
__global__ void prep_weights(const float* __restrict__ Wx, const float* __restrict__ Wh,
                             bf16* __restrict__ wcomb) {
    int tid = blockIdx.x * blockDim.x + threadIdx.x;
    if (tid >= KT_ * 128 * 32) return;
    int lane = tid & 31;
    int frag = tid >> 5;
    int kt = frag / 128, nt = frag % 128;
    int n = lane & 15, khalf = (lane >> 4) * 16;
    int ncol = nt * 16 + n;
    bf16* dst = wcomb + (size_t)frag * FRAG + lane * 16;
    #pragma unroll
    for (int e = 0; e < 16; ++e) {
        int kg = kt * 32 + (khalf >> 1) + e + ((e >= 8) ? 8 : 0);               // 0..575
        float v = (kg < D_) ? Wx[(size_t)kg * G4_ + ncol]
                            : Wh[(size_t)(kg - D_) * G4_ + ncol];
        dst[e] = (bf16)v;
    }
}

__global__ void prep_wd(const float* __restrict__ Wd, bf16* __restrict__ wdB) {
    int tid = blockIdx.x * blockDim.x + threadIdx.x;
    if (tid >= 16 * 16 * 32) return;
    int lane = tid & 31;
    int frag = tid >> 5;
    int kt = frag / 16, nt = frag % 16;
    int n = lane & 15, khalf = (lane >> 4) * 16;
    int ncol = nt * 16 + n;
    bf16* dst = wdB + (size_t)frag * FRAG + lane * 16;
    #pragma unroll
    for (int e = 0; e < 16; ++e) {
        int kg = kt * 32 + (khalf >> 1) + e + ((e >= 8) ? 8 : 0);               // 0..511
        dst[e] = (bf16)Wd[(size_t)kg * F_ + ncol];
    }
}

// =====================  prep: pack observations into A-fragments  =====================
// A layout (16-bit, 16x32): lane = m + 16*hi; lane<16 holds K in {0..7,16..23}, lane>=16 K in {8..15,24..31}
// elem e => K = 8*(lane>>4) + e + (e>=8 ? 8 : 0)
__global__ void prep_obs(const float* __restrict__ obs, bf16* __restrict__ obsA) {
    int tid = blockIdx.x * blockDim.x + threadIdx.x;
    if (tid >= T_ * 16 * 2 * 32) return;
    int lane = tid & 31;
    int frag = tid >> 5;                 // = t*32 + bTile*2 + kt2
    int t = frag >> 5;
    int bTile = (frag >> 1) & 15;
    int kt2 = frag & 1;
    int m = lane & 15;
    int kb = (lane >> 4) * 8;
    int bb = bTile * 16 + m;
    bf16* dst = obsA + (size_t)frag * FRAG + lane * 16;
    #pragma unroll
    for (int e = 0; e < 16; ++e) {
        int k = kb + e + ((e >= 8) ? 8 : 0);
        int d = kt2 * 32 + k;
        dst[e] = (bf16)obs[((size_t)bb * T_ + t) * D_ + d];
    }
}

__global__ void prep_zero(float* __restrict__ c, bf16* __restrict__ hA0) {
    int tid = blockIdx.x * blockDim.x + threadIdx.x;
    if (tid < B_ * H_) { c[tid] = 0.0f; hA0[tid] = (bf16)0.0f; }
}

// =====================  LSTM step: gates = [obs_t|h] @ [Wx;Wh] + b  =====================
// grid (4,16): blockIdx.x = 64-row block, blockIdx.y = 32-hidden-col block. 8 waves/WG.
// wave w: mTile = bx*4 + (w&3); gate pair gp = w>>2 covers gates {2gp, 2gp+1} x 2 jSub tiles.
__global__ __launch_bounds__(256) void lstm_step(
        const bf16* __restrict__ obsA, const bf16* __restrict__ wcomb,
        const float* __restrict__ bias,
        const bf16* __restrict__ hCur, bf16* __restrict__ hNext,
        float* __restrict__ c, int t) {
    __shared__ float gatesS[4 * 64 * 32];          // 32 KB
    const int tid  = threadIdx.x;
    const int lane = tid & 31;
    const int wave = tid >> 5;
    const int mLoc = wave & 3;
    const int gp   = wave >> 2;
    const int mTile = blockIdx.x * 4 + mLoc;

    v8f acc[4];
    #pragma unroll
    for (int i = 0; i < 4; ++i) { v8f z = {0.f,0.f,0.f,0.f,0.f,0.f,0.f,0.f}; acc[i] = z; }

    int nTileG[4];
    #pragma unroll
    for (int i = 0; i < 4; ++i) {
        int q = gp * 2 + (i >> 1);
        int js = i & 1;
        nTileG[i] = q * 32 + blockIdx.y * 2 + js;   // global N-tile over 2048 cols
    }

    const bf16* aObs = obsA + ((size_t)t * 32 + mTile * 2) * FRAG + lane * 16;
    const bf16* aH   = hCur + (size_t)mTile * 16 * FRAG + lane * 16;

    for (int kt = 0; kt < KT_; ++kt) {
        const bf16* ap = (kt < 2) ? (aObs + (size_t)kt * FRAG)
                                  : (aH + (size_t)(kt - 2) * FRAG);
        v16bf a = load_frag(ap);
        #pragma unroll
        for (int i = 0; i < 4; ++i) {
            v16bf bm = load_frag(wcomb + ((size_t)kt * 128 + nTileG[i]) * FRAG + lane * 16);
            acc[i] = __builtin_amdgcn_wmma_f32_16x16x32_bf16(
                         false, a, false, bm, (short)0, acc[i], false, false);
        }
    }

    // bias + activations, write to LDS for cross-gate exchange
    const int n   = lane & 15;
    const int rHi = (lane >> 4) * 8;                // C/D: M = v + 8*(lane>>4)
    #pragma unroll
    for (int i = 0; i < 4; ++i) {
        int q  = gp * 2 + (i >> 1);
        int js = i & 1;
        float bval = bias[q * H_ + blockIdx.y * 32 + js * 16 + n];
        int jj = js * 16 + n;
        #pragma unroll
        for (int v = 0; v < 8; ++v) {
            float x = acc[i][v] + bval;
            float y = (q == 2) ? tanhf(x) : sigmoidf_(x);
            int r = mLoc * 16 + rHi + v;
            gatesS[(q * 64 + r) * 32 + jj] = y;
        }
    }
    __syncthreads();

    // state update: 64 rows x 32 hidden cols = 2048 elems, 8 per thread
    #pragma unroll
    for (int u = 0; u < 8; ++u) {
        int idx = u * 256 + tid;
        int r  = idx >> 5;
        int jj = idx & 31;
        float ig = gatesS[(0 * 64 + r) * 32 + jj];
        float fg = gatesS[(1 * 64 + r) * 32 + jj];
        float gg = gatesS[(2 * 64 + r) * 32 + jj];
        float og = gatesS[(3 * 64 + r) * 32 + jj];
        int bG = blockIdx.x * 64 + r;
        int jG = blockIdx.y * 32 + jj;
        size_t cIdx = (size_t)bG * H_ + jG;
        float cn = fg * c[cIdx] + ig * gg;
        c[cIdx] = cn;
        float hn = og * tanhf(cn);
        // scatter into A-fragment layout for next step
        int bTile = bG >> 4, m = bG & 15;
        int kTile = jG >> 5, kk = jG & 31;
        int hi = (kk >> 3) & 1;
        int lw = m + 16 * hi;
        int e  = (kk & 7) + 8 * (kk >> 4);
        hNext[((size_t)bTile * 16 + kTile) * FRAG + lw * 16 + e] = (bf16)hn;
    }
}

// =====================  head: out = relu(h_last @ Wd + bd)  =====================
// grid (4,2): 64-row x 128-col tile per WG; wave: mTile = bx*4+(w&3), 4 N-tiles.
__global__ __launch_bounds__(256) void dense_head(
        const bf16* __restrict__ hA, const bf16* __restrict__ wdB,
        const float* __restrict__ bd, float* __restrict__ out) {
    const int tid  = threadIdx.x;
    const int lane = tid & 31;
    const int wave = tid >> 5;
    const int mTile = blockIdx.x * 4 + (wave & 3);
    const int nBase = blockIdx.y * 8 + (wave >> 2) * 4;

    v8f acc[4];
    #pragma unroll
    for (int i = 0; i < 4; ++i) { v8f z = {0.f,0.f,0.f,0.f,0.f,0.f,0.f,0.f}; acc[i] = z; }

    for (int kt = 0; kt < 16; ++kt) {
        v16bf a = load_frag(hA + ((size_t)mTile * 16 + kt) * FRAG + lane * 16);
        #pragma unroll
        for (int i = 0; i < 4; ++i) {
            v16bf bm = load_frag(wdB + ((size_t)kt * 16 + nBase + i) * FRAG + lane * 16);
            acc[i] = __builtin_amdgcn_wmma_f32_16x16x32_bf16(
                         false, a, false, bm, (short)0, acc[i], false, false);
        }
    }

    const int n   = lane & 15;
    const int rHi = (lane >> 4) * 8;
    #pragma unroll
    for (int i = 0; i < 4; ++i) {
        int col = (nBase + i) * 16 + n;
        float bval = bd[col];
        #pragma unroll
        for (int v = 0; v < 8; ++v) {
            int row = mTile * 16 + rHi + v;
            float y = acc[i][v] + bval;
            out[(size_t)row * F_ + col] = y > 0.f ? y : 0.f;
        }
    }
}

// =====================  host launcher  =====================
extern "C" void kernel_launch(void* const* d_in, const int* in_sizes, int n_in,
                              void* d_out, int out_size, void* d_ws, size_t ws_size,
                              hipStream_t stream) {
    (void)in_sizes; (void)n_in; (void)out_size; (void)ws_size;
    const float* obs  = (const float*)d_in[0];   // [B,T,D]
    const float* Wx   = (const float*)d_in[1];   // [D,4H]
    const float* Wh   = (const float*)d_in[2];   // [H,4H]
    const float* bias = (const float*)d_in[3];   // [4H]
    const float* Wd   = (const float*)d_in[4];   // [H,F]
    const float* bd   = (const float*)d_in[5];   // [F]
    float* out = (float*)d_out;                  // [B,F] fp32

    char* ws = (char*)d_ws;
    size_t off = 0;
    bf16*  wcomb = (bf16*)(ws + off); off += (size_t)KT_ * 128 * 1024;   // 2.25 MB
    bf16*  wdB   = (bf16*)(ws + off); off += (size_t)16 * 16 * 1024;     // 256 KB
    bf16*  obsA  = (bf16*)(ws + off); off += (size_t)T_ * 32 * 1024;     // 4 MB
    bf16*  hA0   = (bf16*)(ws + off); off += (size_t)256 * 1024;         // 256 KB
    bf16*  hA1   = (bf16*)(ws + off); off += (size_t)256 * 1024;         // 256 KB
    float* cSt   = (float*)(ws + off); off += (size_t)B_ * H_ * 4;       // 512 KB

    prep_weights<<<(KT_ * 128 * 32 + 255) / 256, 256, 0, stream>>>(Wx, Wh, wcomb);
    prep_wd<<<(16 * 16 * 32 + 255) / 256, 256, 0, stream>>>(Wd, wdB);
    prep_obs<<<(T_ * 16 * 2 * 32 + 255) / 256, 256, 0, stream>>>(obs, obsA);
    prep_zero<<<(B_ * H_ + 255) / 256, 256, 0, stream>>>(cSt, hA0);

    for (int t = 0; t < T_; ++t) {
        const bf16* cur = (t & 1) ? hA1 : hA0;
        bf16*       nxt = (t & 1) ? hA0 : hA1;
        lstm_step<<<dim3(4, 16), 256, 0, stream>>>(obsA, wcomb, bias, cur, nxt, cSt, t);
    }
    // T even -> final h in hA0
    dense_head<<<dim3(4, 2), 256, 0, stream>>>(hA0, wdB, bd, out);
}
